// TritonFlashCausalAttention_4492535792219
// MI455X (gfx1250) — matrix operand order repaired and checked by reading, not compiled, not run on hardware
//
#include <hip/hip_runtime.h>
#include <hip/hip_bf16.h>

// ---------------------------------------------------------------------------
// CDNA5 (gfx1250) wave32 WMMA attention block:
//   qkv = x @ Wqkv + bqkv ; flash causal attention w/ ALiBi ; out = o @ Wout + bout
// Matrix math on v_wmma_f32_16x16x32_bf16 (fp32 accumulate).
// All LDS fragment reads are contiguous 16B runs -> ds_load_b128.
// ---------------------------------------------------------------------------

typedef __bf16 v16bf __attribute__((ext_vector_type(16)));
typedef float v8f __attribute__((ext_vector_type(8)));

#define GBM 128
#define GBN 128
#define GBK 32
#define APITCH 40    // A tile row pitch: 32 + 8 pad (bf16 units)
#define BTPITCH 40   // transposed-W tile: [col][k], 32 + 8 pad

// ---------------------------------------------------------------------------
// C[M,N] = A[M,K] @ W[K,N] + bias[N]   (fp32 in/out, bf16 WMMA internally)
// block: 256 threads (8 waves), 128x128 tile; wave = 2x4 grid of 16x16 tiles
// ---------------------------------------------------------------------------
__global__ __launch_bounds__(256)
void gemm_bias_kernel(const float* __restrict__ A, const float* __restrict__ W,
                      const float* __restrict__ bias, float* __restrict__ C,
                      int M, int N, int K)
{
  __shared__ __bf16 Al[GBM * APITCH];   // [row][k]
  __shared__ __bf16 Bt[GBN * BTPITCH];  // [col][k]  (W transposed)

  const int tid  = threadIdx.x;
  const int lane = tid & 31;
  const int wave = tid >> 5;       // 0..7
  const int ln   = lane & 15;
  const int hi   = lane >> 4;

  const int mBlock = blockIdx.y * GBM;
  const int nBlock = blockIdx.x * GBN;
  const int wRow = (wave >> 1) * 32;   // 0,32,64,96
  const int wCol = (wave & 1) * 64;    // 0,64

  v8f acc[2][4];
  #pragma unroll
  for (int i = 0; i < 2; ++i)
    #pragma unroll
    for (int j = 0; j < 4; ++j)
      #pragma unroll
      for (int e = 0; e < 8; ++e) acc[i][j][e] = 0.f;

  for (int kb = 0; kb < K; kb += GBK) {
    { // stage A tile 128x32 row-major: contiguous global read, contiguous LDS write
      int row = tid >> 1;
      int off = (tid & 1) * 16;
      const float* src = A + (size_t)(mBlock + row) * K + kb + off;
      #pragma unroll
      for (int e = 0; e < 16; ++e)
        Al[row * APITCH + off + e] = (__bf16)src[e];
    }
    { // stage W tile 32x128 TRANSPOSED: lanes sweep consecutive cols (coalesced),
      // each thread writes a contiguous 16-element k-run for one column.
      int col = tid & 127;
      int kh  = tid >> 7;    // 0/1 -> k 0..15 / 16..31
      const float* src = W + (size_t)(kb + 16 * kh) * N + nBlock + col;
      #pragma unroll
      for (int kk = 0; kk < 16; ++kk)
        Bt[col * BTPITCH + 16 * kh + kk] = (__bf16)src[(size_t)kk * N];
    }
    __syncthreads();

    // A fragments: lane ln = row; contiguous 16-elt run (ds_load_b128 x2)
    v16bf afrag[2], bfrag[4];
    #pragma unroll
    for (int i = 0; i < 2; ++i) {
      int row = wRow + i * 16 + ln;
      #pragma unroll
      for (int e = 0; e < 16; ++e) {
        int k = 16 * hi + e;
        afrag[i][e] = Al[row * APITCH + k];
      }
    }
    // B fragments: lane ln = column; contiguous 16-elt run (ds_load_b128 x2)
    #pragma unroll
    for (int j = 0; j < 4; ++j) {
      int col = wCol + j * 16 + ln;
      #pragma unroll
      for (int e = 0; e < 16; ++e)
        bfrag[j][e] = Bt[col * BTPITCH + 16 * hi + e];
    }

    #pragma unroll
    for (int i = 0; i < 2; ++i)
      #pragma unroll
      for (int j = 0; j < 4; ++j)
        acc[i][j] = __builtin_amdgcn_wmma_f32_16x16x32_bf16(
            false, afrag[i], false, bfrag[j], (short)0, acc[i][j], false, false);

    __syncthreads();
  }

  // C/D layout: lane -> col = ln, VGPR e -> row = e + 8*hi
  #pragma unroll
  for (int i = 0; i < 2; ++i)
    #pragma unroll
    for (int j = 0; j < 4; ++j) {
      int col = nBlock + wCol + j * 16 + ln;
      float b = bias[col];
      #pragma unroll
      for (int e = 0; e < 8; ++e) {
        int row = mBlock + wRow + i * 16 + 8 * hi + e;
        C[(size_t)row * N + col] = acc[i][j][e] + b;
      }
    }
}

// ---------------------------------------------------------------------------
// Flash causal attention with ALiBi.
// qkv layout: [(b*S+s)*3 + c] * (H*HD) + h*HD + hd   (c = 0:q 1:k 2:v)
// grid: (S/64, B*H), block 128 threads (4 waves); wave owns 16 query rows.
// ---------------------------------------------------------------------------
#define FHD 128
#define FBQ 64
#define FBKV 32
#define KPITCH 136   // K tile [kv][hd]: 128 + 8 pad
#define VTPITCH 40   // V tile TRANSPOSED [hd][kv]: 32 + 8 pad

__global__ __launch_bounds__(128)
void flash_attn_kernel(const float* __restrict__ qkv, float* __restrict__ O,
                       int S, int Hn)
{
  const int tid  = threadIdx.x;
  const int lane = tid & 31;
  const int wave = tid >> 5;   // 0..3
  const int ln   = lane & 15;
  const int hi   = lane >> 4;

  const int bh = blockIdx.y;
  const int b  = bh / Hn;
  const int h  = bh % Hn;
  const int qBlock = blockIdx.x * FBQ;
  const int qRow0  = qBlock + wave * 16;

  const float scale = 0.08838834764831845f;          // 1/sqrt(128)
  const float slope = exp2f(-0.5f * (float)(h + 1)); // ALiBi slope, bias_max=8, H=16

  __shared__ __bf16 Kl[FBKV * KPITCH];   // [kv][hd]
  __shared__ __bf16 Vt[FHD * VTPITCH];   // [hd][kv]  (transposed)
  __shared__ __bf16 Pl[4][16 * 32];      // per-wave C->A layout shuffle buffer

  const size_t bsBase = (size_t)b * S;
  const int Dm = Hn * FHD;            // 2048
  const size_t rowStride = (size_t)3 * Dm;  // fp32 elems between consecutive s

  // Preload Q fragments for this wave's 16 rows: 4 fragments cover hd 0..127
  v16bf qfrag[4];
  {
    const float* qp = qkv + ((bsBase + qRow0 + ln) * 3 + 0) * (size_t)Dm + h * FHD;
    #pragma unroll
    for (int f = 0; f < 4; ++f)
      #pragma unroll
      for (int e = 0; e < 16; ++e) {
        int k = 32 * f + 16 * hi + e;
        qfrag[f][e] = (__bf16)qp[k];
      }
  }

  v8f oacc[8];
  #pragma unroll
  for (int j = 0; j < 8; ++j)
    #pragma unroll
    for (int e = 0; e < 8; ++e) oacc[j][e] = 0.f;

  float mrun[8], lrun[8];
  #pragma unroll
  for (int e = 0; e < 8; ++e) { mrun[e] = -__builtin_inff(); lrun[e] = 0.f; }

  const int kvEnd = qBlock + FBQ;     // causal bound for this query block

  for (int kv = 0; kv < kvEnd; kv += FBKV) {
    { // K: row-major stage (contiguous read + write)
      int r   = tid >> 2;
      int off = (tid & 3) * 32;
      const float* kp = qkv + ((bsBase + kv + r) * 3 + 1) * (size_t)Dm + h * FHD + off;
      #pragma unroll
      for (int e = 0; e < 32; ++e)
        Kl[r * KPITCH + off + e] = (__bf16)kp[e];
    }
    { // V: TRANSPOSED stage; lanes sweep consecutive hd (coalesced global),
      // each thread writes one hd-row of 32 contiguous kv elements.
      const float* vp = qkv + ((bsBase + kv) * 3 + 2) * (size_t)Dm + h * FHD + tid;
      #pragma unroll
      for (int kk = 0; kk < 32; ++kk)
        Vt[tid * VTPITCH + kk] = (__bf16)vp[(size_t)kk * rowStride];
    }
    __syncthreads();

    // ---- S = Q @ K^T : two 16-wide key tiles, 4 WMMAs each over hd ----
    float sc[2][8];
    #pragma unroll
    for (int n0 = 0; n0 < 2; ++n0) {
      v8f sacc;
      #pragma unroll
      for (int e = 0; e < 8; ++e) sacc[e] = 0.f;
      #pragma unroll
      for (int f = 0; f < 4; ++f) {
        v16bf kfrag;  // contiguous 16-elt run -> ds_load_b128 x2
        #pragma unroll
        for (int e = 0; e < 16; ++e)
          kfrag[e] = Kl[(n0 * 16 + ln) * KPITCH + 32 * f + 16 * hi + e];
        sacc = __builtin_amdgcn_wmma_f32_16x16x32_bf16(
            false, qfrag[f], false, kfrag, (short)0, sacc, false, false);
      }
      int t = kv + n0 * 16 + ln;                     // key position
      float bias = slope * (float)(t + 1 - S);       // ALiBi
      #pragma unroll
      for (int e = 0; e < 8; ++e) {
        int q = qRow0 + 8 * hi + e;                  // query position
        float s = sacc[e] * scale + bias;
        sc[n0][e] = (t <= q) ? s : -__builtin_inff();
      }
    }

    // ---- online softmax: reduce across the 16 lanes holding each row ----
    float p0v[8], p1v[8], alpha[8];
    #pragma unroll
    for (int e = 0; e < 8; ++e) {
      float mx = fmaxf(sc[0][e], sc[1][e]);
      #pragma unroll
      for (int w = 1; w < 16; w <<= 1)
        mx = fmaxf(mx, __shfl_xor(mx, w, 32));
      float mnew = fmaxf(mrun[e], mx);
      alpha[e] = __expf(mrun[e] - mnew);
      float p0 = __expf(sc[0][e] - mnew);
      float p1 = __expf(sc[1][e] - mnew);
      float rs = p0 + p1;
      #pragma unroll
      for (int w = 1; w < 16; w <<= 1)
        rs += __shfl_xor(rs, w, 32);
      lrun[e] = lrun[e] * alpha[e] + rs;
      mrun[e] = mnew;
      p0v[e] = p0; p1v[e] = p1;
    }

    // ---- rescale running output ----
    #pragma unroll
    for (int j = 0; j < 8; ++j)
      #pragma unroll
      for (int e = 0; e < 8; ++e) oacc[j][e] *= alpha[e];

    // ---- restage P (C layout) -> A layout through per-wave LDS ----
    #pragma unroll
    for (int e = 0; e < 8; ++e) {
      Pl[wave][(8 * hi + e) * 32 + ln]      = (__bf16)p0v[e];
      Pl[wave][(8 * hi + e) * 32 + 16 + ln] = (__bf16)p1v[e];
    }
    __asm__ volatile("s_wait_dscnt 0" ::: "memory");  // intra-wave LDS RAW

    v16bf pfrag;
    #pragma unroll
    for (int e = 0; e < 16; ++e) {
      int k = 16 * hi + e;
      pfrag[e] = Pl[wave][ln * 32 + k];
    }

    // ---- O += P @ V : 8 WMMAs cover hd 0..127 ----
    #pragma unroll
    for (int j = 0; j < 8; ++j) {
      v16bf vfrag;  // transposed V: contiguous 16-elt run -> ds_load_b128 x2
      #pragma unroll
      for (int e = 0; e < 16; ++e)
        vfrag[e] = Vt[(j * 16 + ln) * VTPITCH + 16 * hi + e];
      oacc[j] = __builtin_amdgcn_wmma_f32_16x16x32_bf16(
          false, pfrag, false, vfrag, (short)0, oacc[j], false, false);
    }
    __syncthreads();
  }

  // ---- normalize and store O in (b, s, h*HD + hd) layout ----
  #pragma unroll
  for (int j = 0; j < 8; ++j)
    #pragma unroll
    for (int e = 0; e < 8; ++e) {
      int q = qRow0 + 8 * hi + e;
      O[(bsBase + q) * (size_t)Dm + h * FHD + j * 16 + ln] = oacc[j][e] / lrun[e];
    }
}

// ---------------------------------------------------------------------------
extern "C" void kernel_launch(void* const* d_in, const int* in_sizes, int n_in,
                              void* d_out, int out_size, void* d_ws, size_t ws_size,
                              hipStream_t stream)
{
  (void)in_sizes; (void)n_in; (void)out_size; (void)ws_size;
  const float* x    = (const float*)d_in[0];
  // d_in[1] (attn_mask) is reproduced analytically inside flash_attn_kernel
  const float* Wqkv = (const float*)d_in[2];
  const float* bqkv = (const float*)d_in[3];
  const float* Wout = (const float*)d_in[4];
  const float* bout = (const float*)d_in[5];
  float* out = (float*)d_out;

  const int Bsz = 2, S = 2048, D = 2048, Hn = 16;
  const int M = Bsz * S;                    // 4096

  float* qkv = (float*)d_ws;                // M * 3D fp32  (~100.7 MB)
  float* o   = qkv + (size_t)M * 3 * D;     // M * D fp32   (~33.6 MB)

  // 1) qkv = x @ Wqkv + bqkv
  gemm_bias_kernel<<<dim3((3 * D) / GBN, M / GBM), dim3(256), 0, stream>>>(
      x, Wqkv, bqkv, qkv, M, 3 * D, D);

  // 2) flash causal attention with ALiBi
  flash_attn_kernel<<<dim3(S / FBQ, Bsz * Hn), dim3(128), 0, stream>>>(qkv, o, S, Hn);

  // 3) out = o @ Wout + bout
  gemm_bias_kernel<<<dim3(D / GBN, M / GBM), dim3(256), 0, stream>>>(
      o, Wout, bout, out, M, D, D);
}
